// EncoderBlock_22643067584908
// MI455X (gfx1250) — matrix operand order repaired and checked by reading, not compiled, not run on hardware
//
#include <hip/hip_runtime.h>
#include <hip/hip_bf16.h>
#include <math.h>

typedef __attribute__((ext_vector_type(16))) __bf16 v16bf;
typedef __attribute__((ext_vector_type(8)))  float  v8f;

#define TPB 256
#define S_LEN  2048
#define DMODEL 1024
#define MB ((size_t)1 << 20)

// ---------------- bf16 helpers ----------------
__device__ __forceinline__ unsigned short f2bf_rn(float f) {
  unsigned int u = __float_as_uint(f);
  u += 0x7FFFu + ((u >> 16) & 1u);
  return (unsigned short)(u >> 16);
}
__device__ __forceinline__ float bf2f(unsigned short s) {
  return __uint_as_float(((unsigned int)s) << 16);
}
__device__ __forceinline__ void split2(float f, unsigned short& hi, unsigned short& lo) {
  hi = f2bf_rn(f);
  lo = f2bf_rn(f - bf2f(hi));
}

__device__ __forceinline__ v8f wmma_bf16(v16bf a, v16bf b, v8f c) {
  return __builtin_amdgcn_wmma_f32_16x16x32_bf16(false, a, false, b, (short)0, c, false, false);
}

union FragU { v16bf v; uint4 q[2]; unsigned short u[16]; };

__device__ __forceinline__ v16bf ld_frag(const unsigned short* p0, const unsigned short* p1) {
  FragU f;
  f.q[0] = *(const uint4*)p0;
  f.q[1] = *(const uint4*)p1;
  return f.v;
}

// ---------------- CDNA5 async global->LDS DMA (ASYNCcnt) ----------------
__device__ __forceinline__ void async_b128(unsigned lds_off, const void* gaddr) {
  asm volatile("global_load_async_to_lds_b128 %0, %1, off"
               :: "v"(lds_off), "v"((unsigned long long)gaddr) : "memory");
}
__device__ __forceinline__ void wait_async6() { asm volatile("s_wait_asynccnt 6" ::: "memory"); }
__device__ __forceinline__ void wait_async0() { asm volatile("s_wait_asynccnt 0" ::: "memory"); }

// ---------------- transpose + split: in f32 [R][C] -> out hi/lo bf16 [C][R] ----------------
__global__ __launch_bounds__(TPB) void transpose_split(
    const float* __restrict__ in, unsigned short* __restrict__ outH,
    unsigned short* __restrict__ outL, int R, int C)
{
  __shared__ unsigned short tH[64][65];
  __shared__ unsigned short tL[64][65];
  const size_t zb = (size_t)blockIdx.z * (size_t)R * C;
  const int r0 = blockIdx.y * 64, c0 = blockIdx.x * 64;
  const int tid = threadIdx.x;
#pragma unroll
  for (int i = 0; i < 16; ++i) {
    int idx = tid + i * TPB;
    int ir = idx >> 6, ic = idx & 63;
    float v = in[zb + (size_t)(r0 + ir) * C + (c0 + ic)];
    unsigned short h, l; split2(v, h, l);
    tH[ic][ir] = h; tL[ic][ir] = l;
  }
  __syncthreads();
#pragma unroll
  for (int i = 0; i < 16; ++i) {
    int idx = tid + i * TPB;
    int orr = idx >> 6, oc = idx & 63;
    const size_t o = zb + (size_t)(c0 + orr) * R + (r0 + oc);
    outH[o] = tH[orr][oc];
    outL[o] = tL[orr][oc];
  }
}

// ---------------- LayerNorm -> split bf16 output ----------------
// mode 0: read f32 input; mode 1: reconstruct from split input
__global__ __launch_bounds__(TPB) void ln_split(
    const float* __restrict__ xf,
    const unsigned short* __restrict__ xh, const unsigned short* __restrict__ xl,
    const float* __restrict__ g, const float* __restrict__ b,
    unsigned short* __restrict__ yh, unsigned short* __restrict__ yl,
    int D, int mode)
{
  __shared__ float red0[8], red1[8];
  const int row = blockIdx.x;
  const size_t base = (size_t)row * D;
  float s = 0.f, s2 = 0.f;
  for (int i = threadIdx.x; i < D; i += TPB) {
    float v = (mode == 0) ? xf[base + i] : (bf2f(xh[base + i]) + bf2f(xl[base + i]));
    s += v; s2 += v * v;
  }
  for (int off = 16; off > 0; off >>= 1) {
    s  += __shfl_down(s, off, 32);
    s2 += __shfl_down(s2, off, 32);
  }
  if ((threadIdx.x & 31) == 0) { red0[threadIdx.x >> 5] = s; red1[threadIdx.x >> 5] = s2; }
  __syncthreads();
  float ts = 0.f, ts2 = 0.f;
#pragma unroll
  for (int w = 0; w < 8; ++w) { ts += red0[w]; ts2 += red1[w]; }
  const float mean = ts / (float)D;
  const float var  = ts2 / (float)D - mean * mean;
  const float rstd = rsqrtf(var + 1e-5f);
  for (int i = threadIdx.x; i < D; i += TPB) {
    float v = (mode == 0) ? xf[base + i] : (bf2f(xh[base + i]) + bf2f(xl[base + i]));
    float y = (v - mean) * rstd * g[i] + b[i];
    unsigned short hh, ll; split2(y, hh, ll);
    yh[base + i] = hh; yl[base + i] = ll;
  }
}

// ---------------- bf16x3 GEMM, pre-split operands, async double-buffered staging ----------------
// A: hi/lo bf16 [M][K].  Bt: hi/lo bf16 [N][K] (pre-transposed).
// ep 0: split out = acc+bias           ep 1: split out = acc+bias+auxf
// ep 2: split out = rec(aux)*silu(acc+bias)
// ep 3: f32  out = acc+bias+rec(aux)   ep 4: f32 out = acc+bias
__global__ __launch_bounds__(TPB) void gemm_bf16x3(
    const unsigned short* __restrict__ Ahi, const unsigned short* __restrict__ Alo,
    const unsigned short* __restrict__ Bth, const unsigned short* __restrict__ Btl,
    const float* __restrict__ bias, const float* __restrict__ auxf,
    const unsigned short* __restrict__ auxh, const unsigned short* __restrict__ auxl,
    unsigned short* __restrict__ Chi, unsigned short* __restrict__ Clo,
    float* __restrict__ Cf, int M, int N, int K, int ep)
{
  __shared__ unsigned short As_hi[2][128 * 40];
  __shared__ unsigned short As_lo[2][128 * 40];
  __shared__ unsigned short Bs_hi[2][64 * 40];
  __shared__ unsigned short Bs_lo[2][64 * 40];

  const int tid = threadIdx.x;
  const int lane = tid & 31, wave = tid >> 5;
  const int l16 = lane & 15, lh = lane >> 4;
  const int wm = wave & 3, wn = wave >> 2;
  const int bm = blockIdx.y * 128, bn = blockIdx.x * 64;

  const v8f vzero = {0.f,0.f,0.f,0.f,0.f,0.f,0.f,0.f};
  v8f acc[2][2];
  acc[0][0] = vzero; acc[0][1] = vzero; acc[1][0] = vzero; acc[1][1] = vzero;

  auto stage = [&](int buf, int k0) {
#pragma unroll
    for (int i = 0; i < 2; ++i) {
      int slot = tid + i * TPB;                  // 0..511
      int row = slot >> 2, sub = slot & 3;       // 4 x 16B = 64B = 32 bf16 per row
      const size_t go = ((size_t)(bm + row) * K + k0) * 2 + sub * 16;
      async_b128((unsigned)(size_t)&As_hi[buf][row * 40] + sub * 16, (const char*)Ahi + go);
      async_b128((unsigned)(size_t)&As_lo[buf][row * 40] + sub * 16, (const char*)Alo + go);
    }
    {
      int row = tid >> 2, sub = tid & 3;         // 64 rows of Bt
      const size_t go = ((size_t)(bn + row) * K + k0) * 2 + sub * 16;
      async_b128((unsigned)(size_t)&Bs_hi[buf][row * 40] + sub * 16, (const char*)Bth + go);
      async_b128((unsigned)(size_t)&Bs_lo[buf][row * 40] + sub * 16, (const char*)Btl + go);
    }
  };

  stage(0, 0);
  const int nk = K >> 5;
  for (int t = 0; t < nk; ++t) {
    const int cur = t & 1;
    if (t + 1 < nk) { stage(cur ^ 1, (t + 1) << 5); wait_async6(); }
    else            { wait_async0(); }
    __syncthreads();

    v16bf a_hi[2], a_lo[2], b_hi[2], b_lo[2];
#pragma unroll
    for (int tt = 0; tt < 2; ++tt) {
      const unsigned short* pah = &As_hi[cur][(wm*32 + tt*16 + l16)*40 + lh*8];
      const unsigned short* pal = &As_lo[cur][(wm*32 + tt*16 + l16)*40 + lh*8];
      a_hi[tt] = ld_frag(pah, pah + 16);
      a_lo[tt] = ld_frag(pal, pal + 16);
      const unsigned short* pbh = &Bs_hi[cur][(wn*32 + tt*16 + l16)*40 + lh*8];
      const unsigned short* pbl = &Bs_lo[cur][(wn*32 + tt*16 + l16)*40 + lh*8];
      b_hi[tt] = ld_frag(pbh, pbh + 16);
      b_lo[tt] = ld_frag(pbl, pbl + 16);
    }
#pragma unroll
    for (int tm = 0; tm < 2; ++tm)
#pragma unroll
      for (int tn = 0; tn < 2; ++tn) {
        acc[tm][tn] = wmma_bf16(a_hi[tm], b_hi[tn], acc[tm][tn]);
        acc[tm][tn] = wmma_bf16(a_hi[tm], b_lo[tn], acc[tm][tn]);
        acc[tm][tn] = wmma_bf16(a_lo[tm], b_hi[tn], acc[tm][tn]);
      }
    __syncthreads();
  }

  // ---- epilogue ----
#pragma unroll
  for (int tm = 0; tm < 2; ++tm)
#pragma unroll
    for (int tn = 0; tn < 2; ++tn) {
      const int n = bn + wn*32 + tn*16 + l16;
      const float bv = bias[n];
#pragma unroll
      for (int r = 0; r < 8; ++r) {
        const int m = bm + wm*32 + tm*16 + r + 8*lh;
        const size_t idx = (size_t)m * N + n;
        float v = acc[tm][tn][r] + bv;
        if (ep == 1)      v += auxf[idx];
        else if (ep == 2) {
          const float u1 = bf2f(auxh[idx]) + bf2f(auxl[idx]);
          v = u1 * (v / (1.f + __expf(-v)));
        } else if (ep == 3) v += bf2f(auxh[idx]) + bf2f(auxl[idx]);
        if (ep >= 3) Cf[idx] = v;
        else { unsigned short hh, ll; split2(v, hh, ll); Chi[idx] = hh; Clo[idx] = ll; }
      }
    }
}

// ---------------- fused differential flash attention + headwise GroupNorm ----------------
// Q/K pre-split [4096][1024]; V pre-transposed+split per batch [1024][2048].
__global__ __launch_bounds__(TPB) void diff_attn(
    const unsigned short* __restrict__ Qh, const unsigned short* __restrict__ Ql,
    const unsigned short* __restrict__ Kh, const unsigned short* __restrict__ Kl,
    const unsigned short* __restrict__ Vth, const unsigned short* __restrict__ Vtl,
    const float* __restrict__ lq1, const float* __restrict__ lk1,
    const float* __restrict__ lq2, const float* __restrict__ lk2,
    const float* __restrict__ gnw, const float* __restrict__ gnb,
    unsigned short* __restrict__ Oh, unsigned short* __restrict__ Ol)
{
  __shared__ unsigned short Ks_hi[2][32][72];
  __shared__ unsigned short Ks_lo[2][32][72];
  __shared__ unsigned short Vt_hi[128][40];
  __shared__ unsigned short Vt_lo[128][40];
  __shared__ unsigned short Ps_hi[8][16][40];
  __shared__ unsigned short Ps_lo[8][16][40];

  const int tid = threadIdx.x;
  const int lane = tid & 31, wave = tid >> 5;
  const int l16 = lane & 15, lh = lane >> 4;
  const int q0 = blockIdx.x * 128;
  const int h  = blockIdx.y;
  const int bb = blockIdx.z;
  const size_t rowbase = (size_t)bb * S_LEN;

  // lambda for this head
  float s1 = 0.f, s2 = 0.f;
  for (int d = 0; d < 64; ++d) {
    s1 += lq1[h*64 + d] * lk1[h*64 + d];
    s2 += lq2[h*64 + d] * lk2[h*64 + d];
  }
  const float lam = __expf(s1) - __expf(s2) + 0.8f;

  // Q fragments straight from pre-split global (two b128 each)
  FragU qfh[2][2], qfl[2][2];
  {
    const int mrow = q0 + wave*16 + l16;
    const size_t qbase = (rowbase + mrow) * DMODEL + h*128;
#pragma unroll
    for (int br = 0; br < 2; ++br)
#pragma unroll
      for (int kw = 0; kw < 2; ++kw) {
        const size_t o = qbase + br*64 + kw*32 + lh*8;
        qfh[br][kw].q[0] = *(const uint4*)(Qh + o);
        qfh[br][kw].q[1] = *(const uint4*)(Qh + o + 16);
        qfl[br][kw].q[0] = *(const uint4*)(Ql + o);
        qfl[br][kw].q[1] = *(const uint4*)(Ql + o + 16);
      }
  }

  const v8f vzero = {0.f,0.f,0.f,0.f,0.f,0.f,0.f,0.f};
  v8f o1[8], o2[8];
#pragma unroll
  for (int e = 0; e < 8; ++e) { o1[e] = vzero; o2[e] = vzero; }
  float mx[2][8], sm[2][8];
#pragma unroll
  for (int br = 0; br < 2; ++br)
#pragma unroll
    for (int r = 0; r < 8; ++r) { mx[br][r] = -1e30f; sm[br][r] = 0.f; }

  for (int kt = 0; kt < S_LEN; kt += 32) {
    __syncthreads();
    // async-stage K (2 br x 32 keys x 64 d) and V^T (128 e x 32 keys)
#pragma unroll
    for (int i = 0; i < 2; ++i) {
      int slot = tid + i * TPB;                       // 0..511
      int br = slot >> 8, n = (slot >> 3) & 31, sub = slot & 7;
      const size_t go = ((rowbase + kt + n) * DMODEL + h*128 + br*64) * 2 + sub * 16;
      async_b128((unsigned)(size_t)&Ks_hi[br][n][0] + sub * 16, (const char*)Kh + go);
      async_b128((unsigned)(size_t)&Ks_lo[br][n][0] + sub * 16, (const char*)Kl + go);
    }
#pragma unroll
    for (int i = 0; i < 2; ++i) {
      int slot = tid + i * TPB;                       // 0..511
      int er = slot >> 2, sub = slot & 3;
      const size_t go = ((size_t)bb * DMODEL * S_LEN + (size_t)(h*128 + er) * S_LEN + kt) * 2 + sub * 16;
      async_b128((unsigned)(size_t)&Vt_hi[er][0] + sub * 16, (const char*)Vth + go);
      async_b128((unsigned)(size_t)&Vt_lo[er][0] + sub * 16, (const char*)Vtl + go);
    }
    wait_async0();
    __syncthreads();

#pragma unroll
    for (int br = 0; br < 2; ++br) {
      v8f s[2]; s[0] = vzero; s[1] = vzero;
#pragma unroll
      for (int nt = 0; nt < 2; ++nt)
#pragma unroll
        for (int kw = 0; kw < 2; ++kw) {
          const unsigned short* pkh = &Ks_hi[br][nt*16 + l16][kw*32 + lh*8];
          const unsigned short* pkl = &Ks_lo[br][nt*16 + l16][kw*32 + lh*8];
          v16bf kh = ld_frag(pkh, pkh + 16);
          v16bf kl = ld_frag(pkl, pkl + 16);
          s[nt] = wmma_bf16(qfh[br][kw].v, kh, s[nt]);
          s[nt] = wmma_bf16(qfh[br][kw].v, kl, s[nt]);
          s[nt] = wmma_bf16(qfl[br][kw].v, kh, s[nt]);
        }
      // online softmax (scale 1/sqrt(64) applied here, exactly)
#pragma unroll
      for (int r = 0; r < 8; ++r) {
        const float sv0 = s[0][r] * 0.125f;
        const float sv1 = s[1][r] * 0.125f;
        float lm = fmaxf(sv0, sv1);
        lm = fmaxf(lm, __shfl_xor(lm, 1, 32));
        lm = fmaxf(lm, __shfl_xor(lm, 2, 32));
        lm = fmaxf(lm, __shfl_xor(lm, 4, 32));
        lm = fmaxf(lm, __shfl_xor(lm, 8, 32));
        const float newm = fmaxf(mx[br][r], lm);
        const float corr = __expf(mx[br][r] - newm);
        mx[br][r] = newm;
        sm[br][r] *= corr;
        if (br == 0) {
#pragma unroll
          for (int e = 0; e < 8; ++e) o1[e][r] *= corr;
        } else {
#pragma unroll
          for (int e = 0; e < 8; ++e) o2[e][r] *= corr;
        }
        const float p0 = __expf(sv0 - newm);
        const float p1 = __expf(sv1 - newm);
        float ps = p0 + p1;
        ps += __shfl_xor(ps, 1, 32);
        ps += __shfl_xor(ps, 2, 32);
        ps += __shfl_xor(ps, 4, 32);
        ps += __shfl_xor(ps, 8, 32);
        sm[br][r] += ps;
        unsigned short hh, ll;
        const int rr = r + 8*lh;
        split2(p0, hh, ll); Ps_hi[wave][rr][l16]      = hh; Ps_lo[wave][rr][l16]      = ll;
        split2(p1, hh, ll); Ps_hi[wave][rr][16 + l16] = hh; Ps_lo[wave][rr][16 + l16] = ll;
      }
      // P @ V (wave-private LDS round-trip)
      const unsigned short* pph = &Ps_hi[wave][l16][lh*8];
      const unsigned short* ppl = &Ps_lo[wave][l16][lh*8];
      v16bf ph = ld_frag(pph, pph + 16);
      v16bf pl = ld_frag(ppl, ppl + 16);
#pragma unroll
      for (int et = 0; et < 8; ++et) {
        const unsigned short* pvh = &Vt_hi[et*16 + l16][lh*8];
        const unsigned short* pvl = &Vt_lo[et*16 + l16][lh*8];
        v16bf vh = ld_frag(pvh, pvh + 16);
        v16bf vl = ld_frag(pvl, pvl + 16);
        if (br == 0) {
          o1[et] = wmma_bf16(ph, vh, o1[et]);
          o1[et] = wmma_bf16(ph, vl, o1[et]);
          o1[et] = wmma_bf16(pl, vh, o1[et]);
        } else {
          o2[et] = wmma_bf16(ph, vh, o2[et]);
          o2[et] = wmma_bf16(ph, vl, o2[et]);
          o2[et] = wmma_bf16(pl, vh, o2[et]);
        }
      }
    }
  }

  // diff combine + per-head GroupNorm + (1-0.8) scale, split bf16 out
#pragma unroll
  for (int r = 0; r < 8; ++r) {
    const float i1 = 1.f / sm[0][r];
    const float i2 = lam / sm[1][r];
    float vals[8];
    float a = 0.f, a2 = 0.f;
#pragma unroll
    for (int et = 0; et < 8; ++et) {
      const float t = o1[et][r] * i1 - o2[et][r] * i2;
      vals[et] = t; a += t; a2 += t * t;
    }
    a  += __shfl_xor(a, 1, 32);  a  += __shfl_xor(a, 2, 32);
    a  += __shfl_xor(a, 4, 32);  a  += __shfl_xor(a, 8, 32);
    a2 += __shfl_xor(a2, 1, 32); a2 += __shfl_xor(a2, 2, 32);
    a2 += __shfl_xor(a2, 4, 32); a2 += __shfl_xor(a2, 8, 32);
    const float mean = a * (1.f / 128.f);
    const float var  = a2 * (1.f / 128.f) - mean * mean;
    const float rstd = rsqrtf(var + 1e-5f);
    const int orow = q0 + wave*16 + r + 8*lh;
    const size_t ob = (rowbase + orow) * DMODEL + h*128;
#pragma unroll
    for (int et = 0; et < 8; ++et) {
      const int e = et*16 + l16;
      const float t = ((vals[et] - mean) * rstd * gnw[h*128 + e] + gnb[h*128 + e]) * (1.0f - 0.8f);
      unsigned short hh, ll; split2(t, hh, ll);
      Oh[ob + e] = hh; Ol[ob + e] = ll;
    }
  }
}

// ---------------- host-side orchestration ----------------
extern "C" void kernel_launch(void* const* d_in, const int* in_sizes, int n_in,
                              void* d_out, int out_size, void* d_ws, size_t ws_size,
                              hipStream_t stream) {
  (void)in_sizes; (void)n_in; (void)out_size; (void)ws_size;
  const float* x    = (const float*)d_in[0];
  const float* ln1g = (const float*)d_in[1];
  const float* ln1b = (const float*)d_in[2];
  const float* ln2g = (const float*)d_in[3];
  const float* ln2b = (const float*)d_in[4];
  const float* Wq   = (const float*)d_in[5];
  const float* bq   = (const float*)d_in[6];
  const float* Wk   = (const float*)d_in[7];
  const float* bk   = (const float*)d_in[8];
  const float* Wv   = (const float*)d_in[9];
  const float* bv   = (const float*)d_in[10];
  const float* Wo   = (const float*)d_in[11];
  const float* bo   = (const float*)d_in[12];
  const float* lq1  = (const float*)d_in[13];
  const float* lk1  = (const float*)d_in[14];
  const float* lq2  = (const float*)d_in[15];
  const float* lk2  = (const float*)d_in[16];
  const float* gnw  = (const float*)d_in[17];
  const float* gnb  = (const float*)d_in[18];
  const float* W1   = (const float*)d_in[19];
  const float* b1   = (const float*)d_in[20];
  const float* Wg   = (const float*)d_in[21];
  const float* bg   = (const float*)d_in[22];
  const float* W2   = (const float*)d_in[23];
  const float* b2   = (const float*)d_in[24];

  char* W = (char*)d_ws;
  typedef unsigned short u16;
  // activations (split bf16 [4096][1024] = 8MB each)
  u16* Hhi = (u16*)(W +   0*MB);  u16* Hlo = (u16*)(W +   8*MB);   // reused as O
  u16* Qhi = (u16*)(W +  16*MB);  u16* Qlo = (u16*)(W +  24*MB);   // reused as X2
  u16* Khi = (u16*)(W +  32*MB);  u16* Klo = (u16*)(W +  40*MB);   // reused as X1
  float* Vf = (float*)(W + 48*MB);                                  // f32 [4096][1024]
  u16* Vth = (u16*)(W +  64*MB);  u16* Vtl = (u16*)(W +  72*MB);
  u16* U1h = (u16*)(W +  80*MB);  u16* U1l = (u16*)(W + 112*MB);    // [4096][4096]
  u16* U2h = (u16*)(W + 144*MB);  u16* U2l = (u16*)(W + 176*MB);
  // pre-transposed split weights
  u16* WqtH = (u16*)(W + 208*MB); u16* WqtL = (u16*)(W + 210*MB);
  u16* WktH = (u16*)(W + 212*MB); u16* WktL = (u16*)(W + 214*MB);
  u16* WvtH = (u16*)(W + 216*MB); u16* WvtL = (u16*)(W + 218*MB);
  u16* WotH = (u16*)(W + 220*MB); u16* WotL = (u16*)(W + 222*MB);
  u16* W1tH = (u16*)(W + 224*MB); u16* W1tL = (u16*)(W + 232*MB);
  u16* WgtH = (u16*)(W + 240*MB); u16* WgtL = (u16*)(W + 272*MB);
  u16* W2tH = (u16*)(W + 304*MB); u16* W2tL = (u16*)(W + 312*MB);
  u16* Ohi = Hhi; u16* Olo = Hlo;
  u16* X1h = Khi; u16* X1l = Klo;
  u16* X2h = Qhi; u16* X2l = Qlo;

  dim3 blk(256);
  // weight prep: W [K][N] -> Wt [N][K] hi/lo
  transpose_split<<<dim3(16, 16, 1), blk, 0, stream>>>(Wq, WqtH, WqtL, 1024, 1024);
  transpose_split<<<dim3(16, 16, 1), blk, 0, stream>>>(Wk, WktH, WktL, 1024, 1024);
  transpose_split<<<dim3(16, 16, 1), blk, 0, stream>>>(Wv, WvtH, WvtL, 1024, 1024);
  transpose_split<<<dim3(16, 16, 1), blk, 0, stream>>>(Wo, WotH, WotL, 1024, 1024);
  transpose_split<<<dim3(64, 16, 1), blk, 0, stream>>>(W1, W1tH, W1tL, 1024, 4096);
  transpose_split<<<dim3(64, 64, 1), blk, 0, stream>>>(Wg, WgtH, WgtL, 4096, 4096);
  transpose_split<<<dim3(16, 64, 1), blk, 0, stream>>>(W2, W2tH, W2tL, 4096, 1024);

  ln_split<<<4096, blk, 0, stream>>>(x, nullptr, nullptr, ln1g, ln1b, Hhi, Hlo, 1024, 0);

  dim3 gD(16, 32), gF(64, 32);
  gemm_bf16x3<<<gD, blk, 0, stream>>>(Hhi, Hlo, WqtH, WqtL, bq, nullptr, nullptr, nullptr,
                                      Qhi, Qlo, nullptr, 4096, 1024, 1024, 0);
  gemm_bf16x3<<<gD, blk, 0, stream>>>(Hhi, Hlo, WktH, WktL, bk, nullptr, nullptr, nullptr,
                                      Khi, Klo, nullptr, 4096, 1024, 1024, 0);
  gemm_bf16x3<<<gD, blk, 0, stream>>>(Hhi, Hlo, WvtH, WvtL, bv, nullptr, nullptr, nullptr,
                                      nullptr, nullptr, Vf, 4096, 1024, 1024, 4);
  // V^T per batch: [2048][1024] -> [1024][2048]
  transpose_split<<<dim3(16, 32, 2), blk, 0, stream>>>(Vf, Vth, Vtl, 2048, 1024);

  diff_attn<<<dim3(16, 8, 2), blk, 0, stream>>>(Qhi, Qlo, Khi, Klo, Vth, Vtl,
                                                lq1, lk1, lq2, lk2, gnw, gnb, Ohi, Olo);

  gemm_bf16x3<<<gD, blk, 0, stream>>>(Ohi, Olo, WotH, WotL, bo, x, nullptr, nullptr,
                                      X1h, X1l, nullptr, 4096, 1024, 1024, 1);
  ln_split<<<4096, blk, 0, stream>>>(nullptr, X1h, X1l, ln2g, ln2b, X2h, X2l, 1024, 1);
  gemm_bf16x3<<<gF, blk, 0, stream>>>(X2h, X2l, W1tH, W1tL, b1, nullptr, nullptr, nullptr,
                                      U1h, U1l, nullptr, 4096, 4096, 1024, 0);
  gemm_bf16x3<<<gF, blk, 0, stream>>>(U1h, U1l, WgtH, WgtL, bg, nullptr, U1h, U1l,
                                      U2h, U2l, nullptr, 4096, 4096, 4096, 2);
  gemm_bf16x3<<<gD, blk, 0, stream>>>(U2h, U2l, W2tH, W2tL, b2, nullptr, X2h, X2l,
                                      nullptr, nullptr, (float*)d_out, 4096, 1024, 4096, 3);
}
